// HyperLinearSSM_90314572300305
// MI455X (gfx1250) — matrix operand order repaired and checked by reading, not compiled, not run on hardware
//
#include <hip/hip_runtime.h>

// Problem dims (match reference)
#define B_ 256
#define T_ 256
#define S_ 64
#define A_ 16
#define H_ 512

typedef __attribute__((ext_vector_type(16))) __bf16 v16bf;
typedef __attribute__((ext_vector_type(8)))  __bf16 v8bf;
typedef __attribute__((ext_vector_type(8)))  float  v8f;

union ABfrag { v16bf v; v8bf h[2]; };

static __device__ __forceinline__ __bf16 f2bf(float f) {
  union { float f; unsigned u; } c; c.f = f;
  unsigned r = c.u + 0x7FFFu + ((c.u >> 16) & 1u);   // round-to-nearest-even
  unsigned short hs = (unsigned short)(r >> 16);
  return __builtin_bit_cast(__bf16, hs);
}

// ---------------------------------------------------------------------------
// One-time: transpose Wa2 [512,4096] and Wb2 [512,1024] into bf16 column-major
// (i.e. [N, K] with K contiguous) so WMMA B-fragments load as 2x b128 chunks.
// ---------------------------------------------------------------------------
__global__ void __launch_bounds__(256)
convert_weights(const float* __restrict__ Wa2, const float* __restrict__ Wb2,
                __bf16* __restrict__ Wa2T, __bf16* __restrict__ Wb2T) {
  const int NA = (S_ * S_) * H_;   // 4096*512
  const int NB = (S_ * A_) * H_;   // 1024*512
  for (int idx = blockIdx.x * blockDim.x + threadIdx.x; idx < NA + NB;
       idx += gridDim.x * blockDim.x) {
    if (idx < NA) {
      int n = idx / H_, k = idx - n * H_;
      Wa2T[idx] = f2bf(Wa2[(size_t)k * (S_ * S_) + n]);
    } else {
      int j = idx - NA;
      int n = j / H_, k = j - n * H_;
      Wb2T[j] = f2bf(Wb2[(size_t)k * (S_ * A_) + n]);
    }
  }
}

// ---------------------------------------------------------------------------
// Precompute v[b,t,:] = B(u_t) @ u_t for ALL (b,t) rows, written into d_out.
// Block = 16 (b,t) rows; 8 waves, each wave owns 8 WMMA N-tiles (128 cols of
// the S*A=1024 hypernet output). Fused: hb = tanh(u@Wb1+bb1) built in LDS,
// then K=512 bf16 WMMA (batched B-frag loads, double-buffered A-frag),
// then per-tile contraction over a (=16 cols) with u.
// ---------------------------------------------------------------------------
__global__ void __launch_bounds__(256)
precompute_v(const float* __restrict__ us, const float* __restrict__ Wb1,
             const float* __restrict__ bb1, const __bf16* __restrict__ Wb2T,
             const float* __restrict__ bb2, float* __restrict__ out) {
  __shared__ float u_lds[16][A_];
  __shared__ __attribute__((aligned(16))) __bf16 hb_lds[16 * 520]; // padded stride

  const int tid = threadIdx.x;
  const int bt0 = blockIdx.x * 16;

  { // load 16x16 u tile
    int r = tid >> 4, a = tid & 15;
    u_lds[r][a] = us[(size_t)(bt0 + r) * A_ + a];
  }
  __syncthreads();

  // hb = tanh(u @ Wb1 + bb1)  -> bf16 LDS tile [16 x 512]
  for (int idx = tid; idx < 16 * H_; idx += 256) {
    int r = idx >> 9, h = idx & (H_ - 1);
    float s = bb1[h];
#pragma unroll
    for (int a = 0; a < A_; ++a) s += u_lds[r][a] * Wb1[a * H_ + h];
    hb_lds[r * 520 + h] = f2bf(tanhf(s));
  }
  __syncthreads();

  const int lane = tid & 31, wv = tid >> 5;
  const int half = lane >> 4, nl = lane & 15;
  const int m = nl;  // A-fragment row = lane % 16

  const __bf16* bbase[8];
#pragma unroll
  for (int tt = 0; tt < 8; ++tt)
    bbase[tt] = Wb2T + (size_t)(wv * 128 + tt * 16 + nl) * H_ + half * 8;

  v8f acc[8];
#pragma unroll
  for (int t = 0; t < 8; ++t)
#pragma unroll
    for (int e = 0; e < 8; ++e) acc[t][e] = 0.f;

  ABfrag af[2];
  af[0].h[0] = *(const v8bf*)&hb_lds[m * 520 + half * 8];
  af[0].h[1] = *(const v8bf*)&hb_lds[m * 520 + half * 8 + 16];

#pragma unroll
  for (int kt = 0; kt < H_ / 32; ++kt) {
    const int cur = kt & 1, nxt = cur ^ 1;
    // batch all 8 B-fragments for this k-slice into distinct registers
    ABfrag bfr[8];
#pragma unroll
    for (int tt = 0; tt < 8; ++tt) {
      const __bf16* bp = bbase[tt] + kt * 32;
      bfr[tt].h[0] = *(const v8bf*)bp;
      bfr[tt].h[1] = *(const v8bf*)(bp + 16);
    }
    if (kt + 1 < H_ / 32) {  // prefetch next A-fragment (LDS)
      const int kb = (kt + 1) * 32 + half * 8;
      af[nxt].h[0] = *(const v8bf*)&hb_lds[m * 520 + kb];
      af[nxt].h[1] = *(const v8bf*)&hb_lds[m * 520 + kb + 16];
    }
#pragma unroll
    for (int tt = 0; tt < 8; ++tt)
      acc[tt] = __builtin_amdgcn_wmma_f32_16x16x32_bf16(
          false, af[cur].v, false, bfr[tt].v, (short)0, acc[tt], false, false);
  }

  // Each 16-wide tile tt is exactly one i (A_ == 16): v[bt,i] = sum_a (C+bb2)*u
#pragma unroll
  for (int tt = 0; tt < 8; ++tt) {
    const int i = wv * 8 + tt;
    const int a = nl;
#pragma unroll
    for (int r = 0; r < 8; ++r) {
      const int mm = r + 8 * half;
      float p = (acc[tt][r] + bb2[i * A_ + a]) * u_lds[mm][a];
      p += __shfl_xor(p, 8, 16);
      p += __shfl_xor(p, 4, 16);
      p += __shfl_xor(p, 2, 16);
      p += __shfl_xor(p, 1, 16);
      if (a == 0) out[(size_t)(bt0 + mm) * S_ + i] = p;
    }
  }
}

// ---------------------------------------------------------------------------
// Per-step: ha = tanh(x @ Wa1 + ba1)  (tiny; VALU fp32, emit bf16)
// ---------------------------------------------------------------------------
__global__ void __launch_bounds__(256)
ha_kernel(const float* __restrict__ x, const float* __restrict__ Wa1,
          const float* __restrict__ ba1, __bf16* __restrict__ ha) {
  const int idx = blockIdx.x * 256 + threadIdx.x;   // over 256*512
  const int b = idx >> 9, h = idx & (H_ - 1);
  float s = ba1[h];
#pragma unroll 8
  for (int j = 0; j < S_; ++j) s += x[b * S_ + j] * Wa1[j * H_ + h];
  ha[(size_t)b * H_ + h] = f2bf(tanhf(s));
}

// ---------------------------------------------------------------------------
// Per-step fused: x_next[b,i] = sum_j (ha@Wa2 + ba2)[b, i*64+j] * x[b,j]
//                              + v[b,t,i]   (v lives in d_out, overwritten)
// Block = 16 b-rows x 8 i-values (one i per wave; 4 N-tiles, K=512 WMMA).
// A- and B-fragments fully double-buffered so WMMAs of iteration kt only
// wait on loads issued an entire iteration earlier.
// ---------------------------------------------------------------------------
__global__ void __launch_bounds__(256)
step_kernel(const float* __restrict__ x, const __bf16* __restrict__ ha,
            const __bf16* __restrict__ Wa2T, const float* __restrict__ ba2,
            float* __restrict__ out, float* __restrict__ xnext, int t) {
  __shared__ float x_lds[16][S_];
  __shared__ __attribute__((aligned(16))) __bf16 ha_lds[16 * 520];

  const int tid = threadIdx.x;
  const int b0 = blockIdx.x * 16;

  for (int idx = tid; idx < 16 * S_; idx += 256) {
    int r = idx >> 6, j = idx & 63;
    x_lds[r][j] = x[(size_t)(b0 + r) * S_ + j];
  }
  for (int idx = tid; idx < 16 * (H_ / 8); idx += 256) { // 8-bf16 chunks
    int r = idx >> 6, c = idx & 63;
    *(v8bf*)&ha_lds[r * 520 + c * 8] =
        *(const v8bf*)&ha[(size_t)(b0 + r) * H_ + c * 8];
  }
  __syncthreads();

  const int lane = tid & 31, wv = tid >> 5;
  const int half = lane >> 4, nl = lane & 15;
  const int i = blockIdx.y * 8 + wv;
  const int m = nl;

  const __bf16* bbase[4];
#pragma unroll
  for (int tt = 0; tt < 4; ++tt)
    bbase[tt] = Wa2T + (size_t)(i * S_ + tt * 16 + nl) * H_ + half * 8;

  v8f acc[4];
#pragma unroll
  for (int tt = 0; tt < 4; ++tt)
#pragma unroll
    for (int e = 0; e < 8; ++e) acc[tt][e] = 0.f;

  ABfrag af[2], bfr[2][4];
  // prologue: fragments for kt = 0
  af[0].h[0] = *(const v8bf*)&ha_lds[m * 520 + half * 8];
  af[0].h[1] = *(const v8bf*)&ha_lds[m * 520 + half * 8 + 16];
#pragma unroll
  for (int tt = 0; tt < 4; ++tt) {
    bfr[0][tt].h[0] = *(const v8bf*)(bbase[tt]);
    bfr[0][tt].h[1] = *(const v8bf*)(bbase[tt] + 16);
  }

#pragma unroll
  for (int kt = 0; kt < H_ / 32; ++kt) {
    const int cur = kt & 1, nxt = cur ^ 1;
    if (kt + 1 < H_ / 32) {  // issue next iteration's loads first
      const int kb = (kt + 1) * 32 + half * 8;
      af[nxt].h[0] = *(const v8bf*)&ha_lds[m * 520 + kb];
      af[nxt].h[1] = *(const v8bf*)&ha_lds[m * 520 + kb + 16];
#pragma unroll
      for (int tt = 0; tt < 4; ++tt) {
        const __bf16* bp = bbase[tt] + (kt + 1) * 32;
        bfr[nxt][tt].h[0] = *(const v8bf*)bp;
        bfr[nxt][tt].h[1] = *(const v8bf*)(bp + 16);
      }
    }
#pragma unroll
    for (int tt = 0; tt < 4; ++tt)
      acc[tt] = __builtin_amdgcn_wmma_f32_16x16x32_bf16(
          false, af[cur].v, false, bfr[cur][tt].v, (short)0, acc[tt], false, false);
  }

  // contract C tiles over j with x, add bias path and v, write state + output
  float p[8];
#pragma unroll
  for (int r = 0; r < 8; ++r) p[r] = 0.f;
#pragma unroll
  for (int tt = 0; tt < 4; ++tt) {
    const int j = tt * 16 + nl;
    const float bias = ba2[i * S_ + j];
#pragma unroll
    for (int r = 0; r < 8; ++r)
      p[r] += (acc[tt][r] + bias) * x_lds[r + 8 * half][j];
  }
#pragma unroll
  for (int r = 0; r < 8; ++r) {
    float s = p[r];
    s += __shfl_xor(s, 8, 16);
    s += __shfl_xor(s, 4, 16);
    s += __shfl_xor(s, 2, 16);
    s += __shfl_xor(s, 1, 16);
    if (nl == 0) {
      const int b = b0 + r + 8 * half;
      const size_t o = ((size_t)b * T_ + t) * S_ + i;
      const float xn = s + out[o];   // out holds precomputed v[b,t,i]
      out[o] = xn;
      xnext[b * S_ + i] = xn;
    }
  }
}

// ---------------------------------------------------------------------------
extern "C" void kernel_launch(void* const* d_in, const int* in_sizes, int n_in,
                              void* d_out, int out_size, void* d_ws, size_t ws_size,
                              hipStream_t stream) {
  const float* x0  = (const float*)d_in[0];
  const float* us  = (const float*)d_in[1];
  const float* Wa1 = (const float*)d_in[2];
  const float* ba1 = (const float*)d_in[3];
  const float* Wa2 = (const float*)d_in[4];
  const float* ba2 = (const float*)d_in[5];
  const float* Wb1 = (const float*)d_in[6];
  const float* bb1 = (const float*)d_in[7];
  const float* Wb2 = (const float*)d_in[8];
  const float* bb2 = (const float*)d_in[9];
  float* out = (float*)d_out;

  char* ws = (char*)d_ws;
  __bf16* Wa2T = (__bf16*)(ws);                                   // 4 MiB
  __bf16* Wb2T = (__bf16*)(ws + (4u << 20));                      // 1 MiB
  __bf16* haB  = (__bf16*)(ws + (5u << 20));                      // 256 KiB
  float*  xb0  = (float*)(ws + (5u << 20) + (512u << 10));        // 64 KiB
  float*  xb1  = (float*)(ws + (5u << 20) + (512u << 10) + (64u << 10));
  (void)in_sizes; (void)n_in; (void)out_size; (void)ws_size;

  convert_weights<<<2048, 256, 0, stream>>>(Wa2, Wb2, Wa2T, Wb2T);
  precompute_v<<<(B_ * T_) / 16, 256, 0, stream>>>(us, Wb1, bb1, Wb2T, bb2, out);

  for (int t = 0; t < T_; ++t) {
    const float* xc = (t == 0) ? x0 : ((t & 1) ? xb0 : xb1);
    float* xn = (t & 1) ? xb1 : xb0;
    ha_kernel<<<(B_ * H_) / 256, 256, 0, stream>>>(xc, Wa1, ba1, haB);
    step_kernel<<<dim3(B_ / 16, S_ / 8), 256, 0, stream>>>(
        xc, haB, Wa2T, ba2, out, xn, t);
  }
}
